// UserRepresentationLayer_66279935312596
// MI455X (gfx1250) — matrix operand order repaired and checked by reading, not compiled, not run on hardware
//
#include <hip/hip_runtime.h>
#include <hip/hip_bf16.h>

typedef __bf16 bf16;
typedef __attribute__((ext_vector_type(16))) __bf16 v16bf;
typedef __attribute__((ext_vector_type(2)))  __bf16 bf16x2;
typedef __attribute__((ext_vector_type(8)))  float  v8f;

#define NB   4096
#define SS   200
#define SP   208     // S padded to 13 tiles of 16
#define DD   64
#define K1   256
#define N1   64
#define N2   32
#define BNEPS 1e-3f

__global__ __launch_bounds__(128, 1)
void din_fused_kernel(const float* __restrict__ em,   const float* __restrict__ eu,
                      const float* __restrict__ Xu,   const float* __restrict__ W1,
                      const float* __restrict__ b1,   const float* __restrict__ W2,
                      const float* __restrict__ b2,   const float* __restrict__ W3,
                      const float* __restrict__ b3,   const float* __restrict__ gamma,
                      const float* __restrict__ beta, const float* __restrict__ mmean,
                      const float* __restrict__ mvar, float* __restrict__ out)
{
    __shared__ __align__(16) bf16  sXu[SP * DD];      // Xu[b] tile, bf16, padded rows zeroed
    __shared__ __align__(16) bf16  sW1t[N1 * K1];     // W1 transposed: [n][k]
    __shared__ __align__(16) bf16  sW2t[N2 * N1];     // W2 transposed: [n][k]
    __shared__ __align__(16) bf16  sEm[DD];
    __shared__ __align__(16) float sEmF[DD];
    __shared__ __align__(16) float sW3[N2];
    __shared__ __align__(16) float sB1[N1];
    __shared__ __align__(16) float sB2[N2];
    __shared__            float sB3;
    __shared__ __align__(16) bf16  sH1[4][16 * N1];   // per-wave GEMM1 output scratch
    __shared__ __align__(16) float sH2[4][16 * N2];   // per-wave GEMM2 output scratch
    __shared__ __align__(16) float sLogits[SP];
    __shared__ __align__(16) float sRed[128];
    __shared__ __align__(16) float sPooled[DD];

    const int b    = blockIdx.x;
    const int tid  = threadIdx.x;
    const int wave = tid >> 5;
    const int lane = tid & 31;
    const int g    = lane >> 4;   // lane group (0: lanes 0-15, 1: lanes 16-31)
    const int lr   = lane & 15;   // row (A-frag) / column (B-frag)

    // ---------------- stage inputs into LDS ----------------
    const float* xb = Xu + (size_t)b * SS * DD;
    for (int i = tid; i < SS * DD; i += 128) sXu[i] = (bf16)xb[i];
    for (int i = SS * DD + tid; i < SP * DD; i += 128) sXu[i] = (bf16)0.0f;
    if (tid < DD) { float e = em[(size_t)b * DD + tid]; sEm[tid] = (bf16)e; sEmF[tid] = e; }
    for (int i = tid; i < N1 * K1; i += 128) {
        int n = i >> 8, k = i & 255;
        sW1t[i] = (bf16)W1[k * N1 + n];
    }
    for (int i = tid; i < N2 * N1; i += 128) {
        int n = i >> 6, k = i & 63;
        sW2t[i] = (bf16)W2[k * N2 + n];
    }
    if (tid < N2) { sW3[tid] = W3[tid]; sB2[tid] = b2[tid]; }
    if (tid < N1) sB1[tid] = b1[tid];
    if (tid == 0) sB3 = b3[0];
    for (int i = tid; i < SP; i += 128) sLogits[i] = -1e30f;
    __syncthreads();

    // ---------------- per-wave M-tiles of 16 sequence rows ----------------
    for (int mt = wave; mt < 13; mt += 4) {
        const int row0 = mt * 16;

        // ---- GEMM1: info[16,256] @ W1[256,64], info built on the fly ----
        v8f c0 = {}, c1 = {}, c2 = {}, c3 = {};
        #pragma unroll
        for (int kt = 0; kt < 8; ++kt) {
            const int quad  = kt >> 1;          // 0:q  1:Xu  2:q-Xu  3:q*Xu
            const int dbase = (kt & 1) * 32;    // which half of D
            v16bf a;
            #pragma unroll
            for (int v = 0; v < 8; ++v) {
                int kl = ((v < 4) ? (2 * v) : (16 + 2 * (v - 4))) + 8 * g;
                int d  = dbase + kl;
                bf16x2 xp = *(const bf16x2*)&sXu[(row0 + lr) * DD + d];
                bf16x2 ep = *(const bf16x2*)&sEm[d];
                bf16 r0, r1;
                if (quad == 0)      { r0 = ep[0]; r1 = ep[1]; }
                else if (quad == 1) { r0 = xp[0]; r1 = xp[1]; }
                else if (quad == 2) { r0 = (bf16)((float)ep[0] - (float)xp[0]);
                                      r1 = (bf16)((float)ep[1] - (float)xp[1]); }
                else                { r0 = (bf16)((float)ep[0] * (float)xp[0]);
                                      r1 = (bf16)((float)ep[1] * (float)xp[1]); }
                a[2 * v] = r0; a[2 * v + 1] = r1;
            }
            #pragma unroll
            for (int nt = 0; nt < 4; ++nt) {
                v16bf bb;
                #pragma unroll
                for (int v = 0; v < 8; ++v) {
                    int k = kt * 32 + ((v < 4) ? (2 * v) : (16 + 2 * (v - 4))) + 8 * g;
                    int n = nt * 16 + lr;
                    bf16x2 wp = *(const bf16x2*)&sW1t[n * K1 + k];
                    bb[2 * v] = wp[0]; bb[2 * v + 1] = wp[1];
                }
                v8f& c = (nt == 0) ? c0 : ((nt == 1) ? c1 : ((nt == 2) ? c2 : c3));
                c = __builtin_amdgcn_wmma_f32_16x16x32_bf16(false, a, false, bb,
                                                            (short)0, c, false, false);
            }
        }

        // ---- bias + relu -> h1 (bf16) in per-wave LDS scratch ----
        bf16* h1 = sH1[wave];
        #pragma unroll
        for (int nt = 0; nt < 4; ++nt) {
            const v8f c = (nt == 0) ? c0 : ((nt == 1) ? c1 : ((nt == 2) ? c2 : c3));
            int n = nt * 16 + lr;
            float bias = sB1[n];
            #pragma unroll
            for (int r = 0; r < 8; ++r) {
                int m = r + 8 * g;
                float h = c[r] + bias; h = h > 0.0f ? h : 0.0f;
                h1[m * N1 + n] = (bf16)h;
            }
        }
        __builtin_amdgcn_wave_barrier();
        asm volatile("s_wait_dscnt 0" ::: "memory");   // per-wave LDS store->load handoff

        // ---- GEMM2: h1[16,64] @ W2[64,32] ----
        v8f d0 = {}, d1 = {};
        #pragma unroll
        for (int kt = 0; kt < 2; ++kt) {
            v16bf a;
            #pragma unroll
            for (int v = 0; v < 8; ++v) {
                int kl = ((v < 4) ? (2 * v) : (16 + 2 * (v - 4))) + 8 * g;
                int k  = kt * 32 + kl;
                bf16x2 hp = *(const bf16x2*)&h1[lr * N1 + k];
                a[2 * v] = hp[0]; a[2 * v + 1] = hp[1];
            }
            #pragma unroll
            for (int nt = 0; nt < 2; ++nt) {
                v16bf bb;
                #pragma unroll
                for (int v = 0; v < 8; ++v) {
                    int k = kt * 32 + ((v < 4) ? (2 * v) : (16 + 2 * (v - 4))) + 8 * g;
                    int n = nt * 16 + lr;
                    bf16x2 wp = *(const bf16x2*)&sW2t[n * N1 + k];
                    bb[2 * v] = wp[0]; bb[2 * v + 1] = wp[1];
                }
                v8f& dd = (nt == 0) ? d0 : d1;
                dd = __builtin_amdgcn_wmma_f32_16x16x32_bf16(false, a, false, bb,
                                                             (short)0, dd, false, false);
            }
        }

        // ---- bias + relu -> h2 (f32) in per-wave LDS scratch ----
        float* h2 = sH2[wave];
        #pragma unroll
        for (int nt = 0; nt < 2; ++nt) {
            const v8f dd = (nt == 0) ? d0 : d1;
            int n = nt * 16 + lr;
            float bias = sB2[n];
            #pragma unroll
            for (int r = 0; r < 8; ++r) {
                int m = r + 8 * g;
                float h = dd[r] + bias; h = h > 0.0f ? h : 0.0f;
                h2[m * N2 + n] = h;
            }
        }
        __builtin_amdgcn_wave_barrier();
        asm volatile("s_wait_dscnt 0" ::: "memory");

        // ---- logits: N=1 projection done as 16-lane dot products ----
        if (g == 0) {
            float acc = sB3;
            #pragma unroll
            for (int n = 0; n < N2; ++n) acc += h2[lr * N2 + n] * sW3[n];
            int s = row0 + lr;
            if (s < SS) sLogits[s] = acc;
        }
    }
    __syncthreads();

    // ---------------- softmax over S (workgroup reduction) ----------------
    float m = -1e30f;
    for (int s = tid; s < SP; s += 128) m = fmaxf(m, sLogits[s]);
    sRed[tid] = m; __syncthreads();
    for (int off = 64; off > 0; off >>= 1) {
        if (tid < off) sRed[tid] = fmaxf(sRed[tid], sRed[tid + off]);
        __syncthreads();
    }
    const float mx = sRed[0];
    __syncthreads();

    float psum = 0.0f;
    for (int s = tid; s < SP; s += 128) {
        float e = __expf(sLogits[s] - mx);   // padded rows: exp(-huge) == 0
        sLogits[s] = e;
        psum += e;
    }
    sRed[tid] = psum; __syncthreads();
    for (int off = 64; off > 0; off >>= 1) {
        if (tid < off) sRed[tid] += sRed[tid + off];
        __syncthreads();
    }
    const float inv = 1.0f / sRed[0];
    __syncthreads();

    // ---------------- pooled = sum_s w_s * Xu[b,s,:] ----------------
    {
        int d = tid & 63, h = tid >> 6;        // 2 threads per feature, half the seq each
        float acc = 0.0f;
        for (int s = h * 100; s < (h + 1) * 100; ++s)
            acc += sLogits[s] * (float)sXu[s * DD + d];
        sRed[tid] = acc * inv;
    }
    __syncthreads();
    if (tid < DD) sPooled[tid] = sRed[tid] + sRed[tid + 64];
    __syncthreads();

    // ---------------- BatchNorm(inference) + concat ----------------
    {
        float x  = (tid < DD) ? sPooled[tid] : sEmF[tid - DD];
        float bn = (x - mmean[tid]) * rsqrtf(mvar[tid] + BNEPS) * gamma[tid] + beta[tid];
        out[(size_t)b * 192 + tid] = bn;
    }
    if (tid < DD) out[(size_t)b * 192 + 128 + tid] = eu[(size_t)b * DD + tid];
}

extern "C" void kernel_launch(void* const* d_in, const int* in_sizes, int n_in,
                              void* d_out, int out_size, void* d_ws, size_t ws_size,
                              hipStream_t stream) {
    (void)in_sizes; (void)n_in; (void)d_ws; (void)ws_size; (void)out_size;
    const float* em    = (const float*)d_in[0];
    const float* eu    = (const float*)d_in[1];
    const float* Xu    = (const float*)d_in[2];
    const float* W1    = (const float*)d_in[3];
    const float* b1    = (const float*)d_in[4];
    const float* W2    = (const float*)d_in[5];
    const float* b2    = (const float*)d_in[6];
    const float* W3    = (const float*)d_in[7];
    const float* b3    = (const float*)d_in[8];
    const float* gamma = (const float*)d_in[9];
    const float* beta  = (const float*)d_in[10];
    const float* mmean = (const float*)d_in[11];
    const float* mvar  = (const float*)d_in[12];
    float* out = (float*)d_out;

    din_fused_kernel<<<NB, 128, 0, stream>>>(em, eu, Xu, W1, b1, W2, b2, W3, b3,
                                             gamma, beta, mmean, mvar, out);
}
